// AttnRNNDecoder_88321707475331
// MI455X (gfx1250) — compile-verified, matrix operand order checked by reading, not run on hardware
//
#include <hip/hip_runtime.h>
#include <hip/hip_bf16.h>

typedef __attribute__((ext_vector_type(16))) __bf16 v16bf;
typedef __attribute__((ext_vector_type(8)))  __bf16 v8bf;
typedef __attribute__((ext_vector_type(4)))  __bf16 v4bf;
typedef __attribute__((ext_vector_type(8)))  float  v8f;

#define NRC 1000
#define BB  64
#define LL  512
#define EE  512
#define HH  512
#define DOD 256
#define VV  100
#define TT  32
#define NTOT (NRC + VV)   // 1100

// ------------------------- reductions -------------------------
__device__ __forceinline__ float wave_max(float v) {
#pragma unroll
  for (int o = 16; o > 0; o >>= 1) v = fmaxf(v, __shfl_down(v, o, 32));
  return v;
}
__device__ __forceinline__ float wave_sum(float v) {
#pragma unroll
  for (int o = 16; o > 0; o >>= 1) v += __shfl_down(v, o, 32);
  return v;
}
__device__ float block_max(float v, float* red) {
  int lane = threadIdx.x & 31, wv = threadIdx.x >> 5;
  int nw = (blockDim.x + 31) >> 5;
  v = wave_max(v);
  if (lane == 0) red[wv] = v;
  __syncthreads();
  if (wv == 0) {
    float x = (lane < nw) ? red[lane] : -INFINITY;
    x = wave_max(x);
    if (lane == 0) red[0] = x;
  }
  __syncthreads();
  float r = red[0];
  __syncthreads();
  return r;
}
__device__ float block_sum(float v, float* red) {
  int lane = threadIdx.x & 31, wv = threadIdx.x >> 5;
  int nw = (blockDim.x + 31) >> 5;
  v = wave_sum(v);
  if (lane == 0) red[wv] = v;
  __syncthreads();
  if (wv == 0) {
    float x = (lane < nw) ? red[lane] : 0.f;
    x = wave_sum(x);
    if (lane == 0) red[0] = x;
  }
  __syncthreads();
  float r = red[0];
  __syncthreads();
  return r;
}

// ------------------------- WMMA GEMM --------------------------
// CDNA5 16-bit A 16x32 tile: lanes 0-15 -> row, k 0..7 & 16..23;
// lanes 16-31 -> row, k 8..15 & 24..31. B loads symmetrically from
// row-major W[N,K] (all GEMMs here are X * W^T).
__device__ __forceinline__ v16bf load_tile(const __bf16* __restrict__ p0,
                                           int ld, int row0, int k0) {
  int lane = threadIdx.x & 31;
  const __bf16* p = p0 + (size_t)(row0 + (lane & 15)) * ld + k0 + ((lane >> 4) << 3);
  v16bf v;
#pragma unroll
  for (int i = 0; i < 8; ++i) v[i] = p[i];
#pragma unroll
  for (int i = 0; i < 8; ++i) v[8 + i] = p[16 + i];
  return v;
}

// D = A[M,K] * W[N,K]^T (+bias)(+relu). One 16x16 tile per wave, 4 waves/block.
// grid.y = M/16, grid.x covers ntiles N-tiles. ncols guards stores (N padding),
// ldc is the output leading dimension. C (f32) and Cb (bf16) each optional.
__global__ void __launch_bounds__(128)
wmma_gemm_bf16(const __bf16* __restrict__ A, const __bf16* __restrict__ W,
               const float* __restrict__ bias,
               float* __restrict__ C, __bf16* __restrict__ Cb,
               int K, int ntiles, int ncols, int ldc, int act) {
  int wave = threadIdx.x >> 5;
  int nt = blockIdx.x * 4 + wave;
  if (nt >= ntiles) return;               // wave-uniform: EXEC stays all-1s for WMMA
  int mt = blockIdx.y;
  v8f acc = {};
  for (int k0 = 0; k0 < K; k0 += 32) {
    v16bf a = load_tile(A, K, mt << 4, k0);
    v16bf b = load_tile(W, K, nt << 4, k0);
    acc = __builtin_amdgcn_wmma_f32_16x16x32_bf16(false, a, false, b,
                                                  (short)0, acc, false, false);
  }
  int lane = threadIdx.x & 31;
  int n = (nt << 4) + (lane & 15);
  if (n >= ncols) return;                 // store-side guard only (post-WMMA)
  int mrow = (mt << 4) + ((lane >> 4) << 3);
  float bv = bias ? bias[n] : 0.f;
#pragma unroll
  for (int i = 0; i < 8; ++i) {
    float v = acc[i] + bv;
    if (act) v = fmaxf(v, 0.f);
    size_t o = (size_t)(mrow + i) * ldc + n;
    if (C)  C[o]  = v;
    if (Cb) Cb[o] = (__bf16)v;
  }
}

// ------------------------- prep kernels -----------------------
__global__ void cvt_bf16_pad(const float* __restrict__ src, __bf16* __restrict__ dst,
                             long n, long ntot) {
  long i = (long)blockIdx.x * blockDim.x + threadIdx.x;
  if (i < ntot) dst[i] = (i < n) ? (__bf16)src[i] : (__bf16)0.f;
}

__global__ void prep_gates(const float* __restrict__ wih, const float* __restrict__ whh,
                           const float* __restrict__ bih, const float* __restrict__ bhh,
                           __bf16* __restrict__ wg, float* __restrict__ bg) {
  long i = (long)blockIdx.x * blockDim.x + threadIdx.x;
  if (i < (long)4 * HH * 2 * HH) {
    int n = (int)(i >> 10);               // 2*HH = 1024 = K
    int k = (int)(i & 1023);
    float v = (k < HH) ? wih[(size_t)n * HH + k] : whh[(size_t)n * HH + (k - HH)];
    wg[i] = (__bf16)v;
  }
  if (i < 4 * HH) bg[i] = bih[i] + bhh[i];
}

__global__ void embed_kernel(const int* __restrict__ idx, const float* __restrict__ common,
                             const float* __restrict__ batched, __bf16* __restrict__ out) {
  int t = blockIdx.x, b = blockIdx.y;
  int id = idx[b * TT + t];
  const float* src;
  if (id < NRC) {
    int ci = id < 0 ? 0 : id;
    src = common + (size_t)ci * DOD;
  } else {
    int bi = id - NRC; if (bi >= VV) bi = VV - 1;
    src = batched + ((size_t)b * VV + bi) * DOD;
  }
  for (int d = threadIdx.x; d < DOD; d += blockDim.x)
    out[((size_t)b * TT + t) * DOD + d] = (__bf16)src[d];
}

__global__ void init_state(float* __restrict__ h, float* __restrict__ c,
                           __bf16* __restrict__ hbf) {
  int i = blockIdx.x * blockDim.x + threadIdx.x;
  if (i < BB * HH) { h[i] = 0.f; c[i] = 0.f; hbf[i] = (__bf16)0.f; }
}

__global__ void concat_bf(const __bf16* __restrict__ x, long xs, int xd,
                          const __bf16* __restrict__ y, long ys, int yd,
                          __bf16* __restrict__ out) {
  int b = blockIdx.x;
  int n = xd + yd;
  for (int i = threadIdx.x; i < n; i += blockDim.x)
    out[(size_t)b * n + i] = (i < xd) ? x[(size_t)b * xs + i]
                                      : y[(size_t)b * ys + (i - xd)];
}

// ------------------- per-step fused kernels -------------------
// scores (b128 row loads) + masked softmax + context (b64 column slabs,
// 4 l-groups x 128 e-quads, LDS partial reduction). This kernel streams the
// two 32 MB bf16 tensors every step and is the per-step bandwidth bottleneck.
__global__ void __launch_bounds__(512)
attn_softmax_kernel(const __bf16* __restrict__ seqproj, const __bf16* __restrict__ enc,
                    const unsigned char* __restrict__ mask,
                    const float* __restrict__ key, __bf16* __restrict__ attn) {
  __shared__ float kf[EE];
  __shared__ float w[LL];
  __shared__ float part[4][EE];
  __shared__ float red[32];
  int b = blockIdx.x;
  for (int e = threadIdx.x; e < EE; e += blockDim.x) kf[e] = key[(size_t)b * EE + e];
  __syncthreads();

  // ---- scores: one row per thread, 16-byte vector loads ----
  int l = threadIdx.x;                     // blockDim == LL == 512
  const v8bf* row = (const v8bf*)(seqproj + ((size_t)b * LL + l) * EE);
  float acc = 0.f;
#pragma unroll 4
  for (int e8 = 0; e8 < EE / 8; ++e8) {
    v8bf v = row[e8];
    const float* kp = kf + e8 * 8;
#pragma unroll
    for (int j = 0; j < 8; ++j) acc += (float)v[j] * kp[j];
  }
  if (!mask[(size_t)b * LL + l]) acc = -INFINITY;

  // ---- softmax ----
  float m = block_max(acc, red);
  float ex = __expf(acc - m);
  float s = block_sum(ex, red);
  w[l] = ex / s;
  __syncthreads();

  // ---- context: attn[e] = sum_l w[l] * enc[l,e] ----
  // 4 groups of 128 threads; group g covers l in [g*128, g*128+128),
  // thread handles 4 consecutive e via one b64 load per row.
  {
    int grp = threadIdx.x >> 7;            // 0..3
    int e4  = threadIdx.x & 127;           // quad index, e = 4*e4
    int l0  = grp * (LL / 4);
    float a0 = 0.f, a1 = 0.f, a2 = 0.f, a3 = 0.f;
    const __bf16* base = enc + (size_t)b * LL * EE + (size_t)l0 * EE + 4 * e4;
    for (int l2 = 0; l2 < LL / 4; ++l2) {
      v4bf p = *(const v4bf*)(base + (size_t)l2 * EE);
      float wl = w[l0 + l2];
      a0 += wl * (float)p[0];
      a1 += wl * (float)p[1];
      a2 += wl * (float)p[2];
      a3 += wl * (float)p[3];
    }
    part[grp][4 * e4 + 0] = a0;
    part[grp][4 * e4 + 1] = a1;
    part[grp][4 * e4 + 2] = a2;
    part[grp][4 * e4 + 3] = a3;
  }
  __syncthreads();
  for (int e = threadIdx.x; e < EE; e += blockDim.x)
    attn[(size_t)b * EE + e] =
        (__bf16)(part[0][e] + part[1][e] + part[2][e] + part[3][e]);
}

__global__ void lstm_pointwise(const float* __restrict__ gates, float* __restrict__ h,
                               float* __restrict__ c, __bf16* __restrict__ hbf) {
  int b = blockIdx.x;
  const float* g = gates + (size_t)b * 4 * HH;
  for (int j = threadIdx.x; j < HH; j += blockDim.x) {
    float ig = 1.f / (1.f + __expf(-g[j]));
    float fg = 1.f / (1.f + __expf(-g[HH + j]));
    float gg = tanhf(g[2 * HH + j]);
    float og = 1.f / (1.f + __expf(-g[3 * HH + j]));
    float cn = fg * c[(size_t)b * HH + j] + ig * gg;
    float hn = og * tanhf(cn);
    c[(size_t)b * HH + j] = cn;
    h[(size_t)b * HH + j] = hn;
    hbf[(size_t)b * HH + j] = (__bf16)hn;
  }
}

// batched logits (per-example matvec, float4 loads) + combined log_softmax
__global__ void __launch_bounds__(256)
final_logits_kernel(const float* __restrict__ outlin, const float* __restrict__ obe,
                    const unsigned char* __restrict__ obemask,
                    float* __restrict__ logits, float* __restrict__ out, int t) {
  __shared__ float red[32];
  __shared__ float xv[DOD];
  int b = blockIdx.x;
  for (int d = threadIdx.x; d < DOD; d += blockDim.x) xv[d] = outlin[(size_t)b * DOD + d];
  __syncthreads();
  for (int v = threadIdx.x; v < VV; v += blockDim.x) {
    const float4* row = (const float4*)(obe + ((size_t)b * VV + v) * DOD);
    float a = 0.f;
#pragma unroll 4
    for (int d4 = 0; d4 < DOD / 4; ++d4) {
      float4 r = row[d4];
      const float* xp = xv + d4 * 4;
      a += r.x * xp[0] + r.y * xp[1] + r.z * xp[2] + r.w * xp[3];
    }
    logits[(size_t)b * NTOT + NRC + v] = obemask[(size_t)b * VV + v] ? a : -INFINITY;
  }
  __syncthreads();
  float lm = -INFINITY;
  for (int j = threadIdx.x; j < NTOT; j += blockDim.x)
    lm = fmaxf(lm, logits[(size_t)b * NTOT + j]);
  float m = block_max(lm, red);
  float ls = 0.f;
  for (int j = threadIdx.x; j < NTOT; j += blockDim.x)
    ls += __expf(logits[(size_t)b * NTOT + j] - m);
  float s = block_sum(ls, red);
  float lse = m + __logf(s);
  float* o = out + ((size_t)b * (TT - 1) + t) * NTOT;
  for (int j = threadIdx.x; j < NTOT; j += blockDim.x)
    o[j] = logits[(size_t)b * NTOT + j] - lse;
}

// --------------------------- host -----------------------------
extern "C" void kernel_launch(void* const* d_in, const int* in_sizes, int n_in,
                              void* d_out, int out_size, void* d_ws, size_t ws_size,
                              hipStream_t stream) {
  const float*         enc      = (const float*)d_in[0];
  const unsigned char* enc_mask = (const unsigned char*)d_in[1];
  const float*         obe      = (const float*)d_in[2];
  const unsigned char* obe_mask = (const unsigned char*)d_in[3];
  const int*           tgt_idx  = (const int*)d_in[4];
  const float* common = (const float*)d_in[5];
  const float* akW = (const float*)d_in[6];  const float* akb = (const float*)d_in[7];
  const float* asW = (const float*)d_in[8];  const float* asb = (const float*)d_in[9];
  const float* cbW = (const float*)d_in[10]; const float* cbb = (const float*)d_in[11];
  const float* wih = (const float*)d_in[12]; const float* whh = (const float*)d_in[13];
  const float* bih = (const float*)d_in[14]; const float* bhh = (const float*)d_in[15];
  const float* owW = (const float*)d_in[16]; const float* owb = (const float*)d_in[17];
  float* out = (float*)d_out;

  char* ws = (char*)d_ws;
  size_t off = 0;
  auto alloc = [&](size_t bytes) -> char* {
    off = (off + 255) & ~(size_t)255;
    char* p = ws + off;
    off += bytes;
    return p;
  };

  __bf16* enc_bf  = (__bf16*)alloc((size_t)BB * LL * EE * 2);       // 32 MB, L2-resident
  __bf16* seqp_bf = (__bf16*)alloc((size_t)BB * LL * EE * 2);       // 32 MB, L2-resident
  __bf16* tgt_bf  = (__bf16*)alloc((size_t)BB * TT * DOD * 2);
  __bf16* wseq    = (__bf16*)alloc((size_t)EE * EE * 2);
  __bf16* wkey    = (__bf16*)alloc((size_t)EE * (HH + DOD) * 2);
  __bf16* wcomb   = (__bf16*)alloc((size_t)HH * (DOD + EE) * 2);
  __bf16* wgate   = (__bf16*)alloc((size_t)4 * HH * 2 * HH * 2);
  float*  bgate   = (float*) alloc((size_t)4 * HH * 4);
  __bf16* wout    = (__bf16*)alloc((size_t)DOD * HH * 2);
  __bf16* wcom    = (__bf16*)alloc((size_t)1008 * DOD * 2);         // N padded to 63 tiles
  float*  hbuf    = (float*) alloc((size_t)BB * HH * 4);
  float*  cbuf    = (float*) alloc((size_t)BB * HH * 4);
  __bf16* hbf     = (__bf16*)alloc((size_t)BB * HH * 2);
  __bf16* akf     = (__bf16*)alloc((size_t)BB * (DOD + HH) * 2);
  float*  keyf    = (float*) alloc((size_t)BB * EE * 4);
  __bf16* attnbf  = (__bf16*)alloc((size_t)BB * EE * 2);
  __bf16* caf     = (__bf16*)alloc((size_t)BB * (DOD + EE) * 2);
  __bf16* combbf  = (__bf16*)alloc((size_t)BB * HH * 2);
  __bf16* xg      = (__bf16*)alloc((size_t)BB * 2 * HH * 2);
  float*  gates   = (float*) alloc((size_t)BB * 4 * HH * 4);
  float*  outlin  = (float*) alloc((size_t)BB * DOD * 4);
  __bf16* outlbf  = (__bf16*)alloc((size_t)BB * DOD * 2);
  float*  logits  = (float*) alloc((size_t)BB * NTOT * 4);
  (void)ws_size; (void)in_sizes; (void)n_in; (void)out_size;

  auto g1 = [](long n) { return dim3((unsigned)((n + 255) / 256)); };

  // one-time conversions (fp32 -> bf16) — halves per-step streamed bytes
  cvt_bf16_pad<<<g1((long)BB * LL * EE), 256, 0, stream>>>(enc, enc_bf, (long)BB * LL * EE, (long)BB * LL * EE);
  cvt_bf16_pad<<<g1((long)EE * EE), 256, 0, stream>>>(asW, wseq, (long)EE * EE, (long)EE * EE);
  cvt_bf16_pad<<<g1((long)EE * (HH + DOD)), 256, 0, stream>>>(akW, wkey, (long)EE * (HH + DOD), (long)EE * (HH + DOD));
  cvt_bf16_pad<<<g1((long)HH * (DOD + EE)), 256, 0, stream>>>(cbW, wcomb, (long)HH * (DOD + EE), (long)HH * (DOD + EE));
  cvt_bf16_pad<<<g1((long)DOD * HH), 256, 0, stream>>>(owW, wout, (long)DOD * HH, (long)DOD * HH);
  cvt_bf16_pad<<<g1((long)1008 * DOD), 256, 0, stream>>>(common, wcom, (long)NRC * DOD, (long)1008 * DOD);
  prep_gates<<<g1((long)4 * HH * 2 * HH), 256, 0, stream>>>(wih, whh, bih, bhh, wgate, bgate);
  embed_kernel<<<dim3(TT, BB), 256, 0, stream>>>(tgt_idx, common, obe, tgt_bf);
  init_state<<<g1((long)BB * HH), 256, 0, stream>>>(hbuf, cbuf, hbf);

  // one-time seq_proj GEMM: [B*L,512] x [512,512]^T -> bf16
  wmma_gemm_bf16<<<dim3(8, (BB * LL) / 16), 128, 0, stream>>>(
      enc_bf, wseq, asb, nullptr, seqp_bf, EE, EE / 16, EE, EE, 0);

  for (int t = 0; t < TT - 1; ++t) {
    const __bf16* prev = tgt_bf + (size_t)t * DOD;   // per-b stride TT*DOD
    // attn_key = [prev_enc, h] @ attn_key_W^T + b
    concat_bf<<<BB, 256, 0, stream>>>(prev, (long)TT * DOD, DOD, hbf, HH, HH, akf);
    wmma_gemm_bf16<<<dim3(8, BB / 16), 128, 0, stream>>>(
        akf, wkey, akb, keyf, nullptr, DOD + HH, EE / 16, EE, EE, 0);
    // scores + softmax + context
    attn_softmax_kernel<<<BB, 512, 0, stream>>>(seqp_bf, enc_bf, enc_mask, keyf, attnbf);
    // comb = relu([prev_enc, attn] @ comb_W^T + b)
    concat_bf<<<BB, 256, 0, stream>>>(prev, (long)TT * DOD, DOD, attnbf, EE, EE, caf);
    wmma_gemm_bf16<<<dim3(8, BB / 16), 128, 0, stream>>>(
        caf, wcomb, cbb, nullptr, combbf, DOD + EE, HH / 16, HH, HH, 1);
    // gates = [comb, h] @ [W_ih|W_hh]^T + (b_ih + b_hh)
    concat_bf<<<BB, 256, 0, stream>>>(combbf, HH, HH, hbf, HH, HH, xg);
    wmma_gemm_bf16<<<dim3(32, BB / 16), 128, 0, stream>>>(
        xg, wgate, bgate, gates, nullptr, 2 * HH, (4 * HH) / 16, 4 * HH, 4 * HH, 0);
    lstm_pointwise<<<BB, 256, 0, stream>>>(gates, hbuf, cbuf, hbf);
    // out_lin = h @ out_W^T + b
    wmma_gemm_bf16<<<dim3(4, BB / 16), 128, 0, stream>>>(
        hbf, wout, owb, outlin, outlbf, HH, DOD / 16, DOD, DOD, 0);
    // proj_common = out_lin @ common_emb^T  (into logits[:, 0:1000], ldc=1100)
    wmma_gemm_bf16<<<dim3(16, BB / 16), 128, 0, stream>>>(
        outlbf, wcom, nullptr, logits, nullptr, DOD, 63, NRC, NTOT, 0);
    // proj_batched + log_softmax + write out[:, t, :]
    final_logits_kernel<<<BB, 256, 0, stream>>>(outlin, obe, obe_mask, logits, out, t);
  }
}
